// RAMENLayer_80178449482251
// MI455X (gfx1250) — matrix operand order repaired and checked by reading, not compiled, not run on hardware
//
#include <hip/hip_runtime.h>
#include <hip/hip_bf16.h>
#include <cstdint>
#include <cstddef>

typedef __bf16 bf16;
typedef __attribute__((ext_vector_type(16))) __bf16 v16bf;
typedef __attribute__((ext_vector_type(8)))  __bf16 v8bf;
typedef __attribute__((ext_vector_type(8)))  float  v8f;
typedef int v4i_b128 __attribute__((vector_size(16)));  // matches builtin param type

#define T_LEN 2048
#define XDIM  2048
#define FFD   2816

#if __has_builtin(__builtin_amdgcn_global_load_async_to_lds_b128)
#define HAVE_ASYNC 1
#else
#define HAVE_ASYNC 0
#endif

// ---------- small helpers ----------
__device__ __forceinline__ bf16 f2bf(float f) {
    unsigned u = __builtin_bit_cast(unsigned, f);
    unsigned r = u + 0x7FFFu + ((u >> 16) & 1u);
    unsigned short h = (unsigned short)(r >> 16);
    return __builtin_bit_cast(bf16, h);
}
__device__ __forceinline__ float bf2f(bf16 b) {
    unsigned short s = __builtin_bit_cast(unsigned short, b);
    unsigned u = ((unsigned)s) << 16;
    return __builtin_bit_cast(float, u);
}
__device__ __forceinline__ float sigf(float x) { return 1.f / (1.f + __expf(-x)); }

// 16-byte global->LDS copy: async on CDNA5 (ASYNCcnt), sync fallback otherwise
__device__ __forceinline__ void async_cp16(const bf16* g, bf16* lds) {
#if HAVE_ASYNC
    __builtin_amdgcn_global_load_async_to_lds_b128(
        (__attribute__((address_space(1))) v4i_b128*)(void*)(g),
        (__attribute__((address_space(3))) v4i_b128*)(void*)(lds),
        0, 0);
#else
    *(uint4*)lds = *(const uint4*)g;
#endif
}
__device__ __forceinline__ void wait_async() {
#if HAVE_ASYNC
#if __has_builtin(__builtin_amdgcn_s_wait_asynccnt)
    __builtin_amdgcn_s_wait_asynccnt(0);
#else
    asm volatile("s_wait_asynccnt 0x0" ::: "memory");
#endif
#endif
}

// ---------- elementwise convert / transpose-convert ----------
__global__ void cvt_kernel(const float* __restrict__ in, bf16* __restrict__ out, size_t n) {
    size_t i = (size_t)blockIdx.x * blockDim.x + threadIdx.x;
    if (i < n) out[i] = f2bf(in[i]);
}
__global__ void tconv_kernel(const float* __restrict__ in, bf16* __restrict__ out, int R, int C) {
    int i = blockIdx.x * blockDim.x + threadIdx.x;
    if (i < R * C) {
        int r = i / C, c = i - r * C;
        out[(size_t)c * R + r] = f2bf(in[i]);
    }
}

// ---------- scale-norm: out_bf16 = x * g / (||x|| + 1e-8) ----------
__global__ __launch_bounds__(256) void norm_kernel(const float* __restrict__ x,
                                                   const float* __restrict__ g,
                                                   bf16* __restrict__ o, int D) {
    const int row = blockIdx.x;
    __shared__ float red[8];
    __shared__ float scale_s;
    float s = 0.f;
    for (int c = threadIdx.x; c < D; c += 256) {
        float v = x[(size_t)row * D + c];
        s += v * v;
    }
    for (int off = 16; off > 0; off >>= 1) s += __shfl_xor(s, off, 32);
    if ((threadIdx.x & 31) == 0) red[threadIdx.x >> 5] = s;
    __syncthreads();
    if (threadIdx.x == 0) {
        float t = 0.f;
        for (int i = 0; i < 8; ++i) t += red[i];
        scale_s = (*g) / (sqrtf(t) + 1e-8f);
    }
    __syncthreads();
    const float sc = scale_s;
    for (int c = threadIdx.x; c < D; c += 256)
        o[(size_t)row * D + c] = f2bf(x[(size_t)row * D + c] * sc);
}

// ---------- WMMA bf16 GEMM: out = epilogue(A @ W^T) ----------
// A: M x K (bf16 row-major), W: N x K (bf16 row-major)
// mode: 0=none 1=sigmoid 2=silu; then *gate, +resid; store f32 and/or bf16
#define BK 32
#define LDA 48  // padded LDS row stride (bf16); 96B -> all frag reads 16B aligned

__device__ __forceinline__ v16bf fragA(const bf16* base, int lane) {
    int r = lane & 15;
    int kb = (lane >> 4) * 8;  // lanes 0-15: K{0..7,16..23}; lanes 16-31: K{8..15,24..31}
    const bf16* p = base + r * LDA + kb;
    v8bf lo = *(const v8bf*)(p);
    v8bf hi = *(const v8bf*)(p + 16);
    return __builtin_shufflevector(lo, hi, 0,1,2,3,4,5,6,7,8,9,10,11,12,13,14,15);
}
__device__ __forceinline__ v16bf fragB(const bf16* base, int lane) {
    int n = lane & 15;
    int kb = (lane >> 4) * 8;  // lanes 0-15: K{0..7,16..23}; lanes 16-31: K{8..15,24..31} of column n
    const bf16* p = base + n * LDA + kb;
    v8bf lo = *(const v8bf*)(p);
    v8bf hi = *(const v8bf*)(p + 16);
    return __builtin_shufflevector(lo, hi, 0,1,2,3,4,5,6,7,8,9,10,11,12,13,14,15);
}

template <int BMt, int BNt, int WR, int WC>
__global__ __launch_bounds__(256) void gemm_bf16_kernel(
    const bf16* __restrict__ A, const bf16* __restrict__ W,
    float* __restrict__ outF, bf16* __restrict__ outB,
    const float* __restrict__ gate, const float* __restrict__ resid,
    int M, int N, int K, int mode) {
    static_assert(WR * WC == 8, "8 waves");
    constexpr int WTM = BMt / WR;          // wave tile M
    constexpr int WTN = BNt / WC;          // wave tile N
    constexpr int MT = WTM / 16;
    constexpr int NT = WTN / 16;
    constexpr int SEGS = (BMt + BNt) / 64; // 16B segments per thread per tile
    static_assert((BMt + BNt) * 4 % 256 == 0, "loader balance");

    __shared__ bf16 smem[2][(BMt + BNt) * LDA];  // A rows [0,BMt), W rows [BMt,BMt+BNt)

    const int tid = threadIdx.x;
    const int lane = tid & 31;
    const int wid = tid >> 5;
    const int bm = blockIdx.y * BMt;
    const int bn = blockIdx.x * BNt;
    const int wm = (wid / WC) * WTM;
    const int wn = (wid % WC) * WTN;

    v8f zero;
    for (int e = 0; e < 8; ++e) zero[e] = 0.f;
    v8f acc[MT][NT];
#pragma unroll
    for (int mt = 0; mt < MT; ++mt)
#pragma unroll
        for (int nt = 0; nt < NT; ++nt) acc[mt][nt] = zero;

    // per-thread staging assignments (16B each)
    const bf16* gp[SEGS];
    int loff[SEGS];
#pragma unroll
    for (int i = 0; i < SEGS; ++i) {
        int s = i * 256 + tid;
        int row = s >> 2;
        int col = (s & 3) * 8;
        if (row < BMt) {
            gp[i] = A + (size_t)(bm + row) * K + col;
            loff[i] = row * LDA + col;
        } else {
            int r2 = row - BMt;
            gp[i] = W + (size_t)(bn + r2) * K + col;
            loff[i] = (BMt + r2) * LDA + col;
        }
    }
    const int nk = K / BK;

    // prologue: stage tile 0
#pragma unroll
    for (int i = 0; i < SEGS; ++i) async_cp16(gp[i], &smem[0][loff[i]]);
    wait_async();
    __syncthreads();

    int cur = 0;
    for (int kt = 0; kt < nk; ++kt) {
        const int nxt = kt + 1;
        if (nxt < nk) {  // overlap next-tile DMA with WMMA on current tile
#pragma unroll
            for (int i = 0; i < SEGS; ++i)
                async_cp16(gp[i] + (size_t)nxt * BK, &smem[cur ^ 1][loff[i]]);
        }
        v16bf afr[MT], bfr[NT];
#pragma unroll
        for (int mt = 0; mt < MT; ++mt)
            afr[mt] = fragA(&smem[cur][(wm + mt * 16) * LDA], lane);
#pragma unroll
        for (int nt = 0; nt < NT; ++nt)
            bfr[nt] = fragB(&smem[cur][(BMt + wn + nt * 16) * LDA], lane);
#pragma unroll
        for (int mt = 0; mt < MT; ++mt)
#pragma unroll
            for (int nt = 0; nt < NT; ++nt)
                acc[mt][nt] = __builtin_amdgcn_wmma_f32_16x16x32_bf16(
                    false, afr[mt], false, bfr[nt], (short)0, acc[mt][nt], false, false);
        if (nxt < nk) {
            wait_async();
            __syncthreads();
            cur ^= 1;
        }
    }

    // epilogue: C 16x16 layout -> row = base + vgpr + 8*(lane>>4), col = base + (lane&15)
    const int colBase = bn + wn + (lane & 15);
    const int rowHalf = (lane >> 4) * 8;
#pragma unroll
    for (int mt = 0; mt < MT; ++mt) {
#pragma unroll
        for (int nt = 0; nt < NT; ++nt) {
            const int col = colBase + nt * 16;
#pragma unroll
            for (int v = 0; v < 8; ++v) {
                const int row = bm + wm + mt * 16 + rowHalf + v;
                float x = acc[mt][nt][v];
                if (mode == 1) x = sigf(x);
                else if (mode == 2) x = x * sigf(x);
                const size_t idx = (size_t)row * N + col;
                if (gate) x *= gate[idx];
                if (resid) x += resid[idx];
                if (outF) outF[idx] = x;
                if (outB) outB[idx] = f2bf(x);
            }
        }
    }
}

// ---------- RoPE precompute ----------
// q_rot: positions t%64+64 ; k_lo: t%64 ; k_hi: t%64+64
__global__ void rope_kernel(const float* __restrict__ q, const float* __restrict__ k,
                            float* __restrict__ qr, float* __restrict__ klo,
                            float* __restrict__ khi) {
    const int idx = blockIdx.x * blockDim.x + threadIdx.x;
    const int pairs = 128;
    if (idx >= T_LEN * pairs) return;
    const int t = idx / pairs;
    const int i = idx - t * pairs;
    const float invf = __expf(-(float)(2 * i) * (9.210340371976184f / 256.f)); // 10000^{-2i/256}
    const float pk = (float)(t & 63);
    const size_t b = (size_t)t * 256 + 2 * i;
    float s, c;
    const float q0 = q[b], q1 = q[b + 1];
    __sincosf((pk + 64.f) * invf, &s, &c);
    qr[b] = q0 * c - q1 * s;
    qr[b + 1] = q1 * c + q0 * s;
    const float k0 = k[b], k1 = k[b + 1];
    khi[b] = k0 * c - k1 * s;
    khi[b + 1] = k1 * c + k0 * s;
    __sincosf(pk * invf, &s, &c);
    klo[b] = k0 * c - k1 * s;
    klo[b + 1] = k1 * c + k0 * s;
}

// ---------- chunked attention: scores+softmax (A to global) ----------
__global__ __launch_bounds__(256) void attn_scores_kernel(const float* __restrict__ qr,
                                                          const float* __restrict__ klo,
                                                          const float* __restrict__ khi,
                                                          float* __restrict__ Aout) {
    const int cIdx = blockIdx.x;        // chunk
    const int tid = threadIdx.x;
    const int lane = tid & 31;
    const int wid = tid >> 5;           // wave -> 8 rows each
    __shared__ float sl[8][128];
    for (int i8 = 0; i8 < 8; ++i8) {
        const int i = wid * 8 + i8;
        float qreg[8];
        for (int dd = 0; dd < 8; ++dd)
            qreg[dd] = qr[(size_t)(cIdx * 64 + i) * 256 + lane + 32 * dd];
        for (int j = 0; j < 128; ++j) {
            const int grow = (cIdx - 1) * 64 + j;  // global row of window entry j
            float partial = 0.f;
            if (grow >= 0) {
                const float* kp = ((j < 64) ? klo : khi) + (size_t)grow * 256;
                for (int dd = 0; dd < 8; ++dd)
                    partial += qreg[dd] * kp[lane + 32 * dd];
            }
            for (int off = 16; off > 0; off >>= 1) partial += __shfl_xor(partial, off, 32);
            float logit = (grow >= 0) ? partial * (1.f / 16.f) : 0.f;  // zero-k rows of chunk 0
            if (j > i + 64) logit = -1e30f;                            // causal window mask
            if (lane == 0) sl[wid][j] = logit;
        }
        __syncthreads();
        float mx = -1e30f;
        for (int j = lane; j < 128; j += 32) mx = fmaxf(mx, sl[wid][j]);
        for (int off = 16; off > 0; off >>= 1) mx = fmaxf(mx, __shfl_xor(mx, off, 32));
        float sm = 0.f;
        for (int j = lane; j < 128; j += 32) sm += __expf(sl[wid][j] - mx);
        for (int off = 16; off > 0; off >>= 1) sm += __shfl_xor(sm, off, 32);
        const float inv = 1.f / sm;
        for (int j = lane; j < 128; j += 32)
            Aout[((size_t)cIdx * 64 + i) * 128 + j] = __expf(sl[wid][j] - mx) * inv;
        __syncthreads();
    }
}

// ---------- y = A @ v2 (per chunk, 256-wide column tiles) ----------
__global__ __launch_bounds__(256) void attn_y_kernel(const float* __restrict__ Abuf,
                                                     const float* __restrict__ V,
                                                     bf16* __restrict__ Y) {
    const int c = blockIdx.y;
    const int col = blockIdx.x * 256 + threadIdx.x;
    __shared__ float Al[64][128];
    for (int idx = threadIdx.x; idx < 64 * 128; idx += 256)
        ((float*)Al)[idx] = Abuf[(size_t)c * 64 * 128 + idx];
    __syncthreads();
    for (int ig = 0; ig < 8; ++ig) {
        float acc[8];
        for (int ii = 0; ii < 8; ++ii) acc[ii] = 0.f;
        for (int j = 0; j < 128; ++j) {
            const int row = (c - 1) * 64 + j;
            float v = 0.f;
            if (row >= 0) v = V[(size_t)row * XDIM + col];
#pragma unroll
            for (int ii = 0; ii < 8; ++ii) acc[ii] += Al[ig * 8 + ii][j] * v;
        }
        for (int ii = 0; ii < 8; ++ii)
            Y[(size_t)(c * 64 + ig * 8 + ii) * XDIM + col] = f2bf(acc[ii]);
    }
}

// ---------- memory recurrence: state[m,d] over t ----------
__global__ __launch_bounds__(256) void scan_kernel(const float* __restrict__ F,
                                                   const float* __restrict__ Vm,
                                                   const float* __restrict__ beta,
                                                   const float* __restrict__ decay,
                                                   bf16* __restrict__ mems) {
    const int m = blockIdx.x;    // 64 slots
    const int d = threadIdx.x;   // 256 channels
    const float sb = sigf(beta[m * 256 + d]);
    const float g = (d < 192) ? 1.f : sigf(decay[m * 64 + (d - 192)]);
    float state = 0.f;
    for (int t = 0; t < T_LEN; ++t) {
        const float f = F[(size_t)t * 64 + m];
        state = (1.f - f * sb) * g * state + f * Vm[(size_t)t * 256 + d];
        mems[((size_t)t * 64 + m) * 256 + d] = f2bf(state);
    }
}

// ---------- memory attend: logits = (mems.qk + l.q)/16, softmax, y = A.mems ----------
__global__ __launch_bounds__(256) void mem_attend_kernel(const bf16* __restrict__ mems,
                                                         const float* __restrict__ qk,
                                                         const float* __restrict__ qm,
                                                         const float* __restrict__ l,
                                                         bf16* __restrict__ y) {
    const int t = blockIdx.x;
    const int tid = threadIdx.x;
    __shared__ float qks[256], qs[256], part[256], logits[64];
    qks[tid] = qk[(size_t)t * 256 + tid];
    qs[tid] = qm[(size_t)t * 256 + tid];
    __syncthreads();
    const int m = tid >> 2;
    const int p = tid & 3;
    float acc = 0.f;
    const bf16* mp = mems + ((size_t)t * 64 + m) * 256;
    const float* lp = l + (size_t)m * 256;
    for (int d = p * 64; d < p * 64 + 64; ++d)
        acc += bf2f(mp[d]) * qks[d] + lp[d] * qs[d];
    part[tid] = acc;
    __syncthreads();
    if (tid < 64)
        logits[tid] = (part[tid * 4] + part[tid * 4 + 1] + part[tid * 4 + 2] + part[tid * 4 + 3]) * (1.f / 16.f);
    __syncthreads();
    float mx = -1e30f;
    for (int i = 0; i < 64; ++i) mx = fmaxf(mx, logits[i]);
    float sm = 0.f;
    for (int i = 0; i < 64; ++i) sm += __expf(logits[i] - mx);
    const float inv = 1.f / sm;
    float yd = 0.f;
    for (int mm = 0; mm < 64; ++mm)
        yd += __expf(logits[mm] - mx) * bf2f(mems[((size_t)t * 64 + mm) * 256 + tid]);
    y[(size_t)t * 256 + tid] = f2bf(yd * inv);
}

// ---------- host launcher ----------
extern "C" void kernel_launch(void* const* d_in, const int* in_sizes, int n_in,
                              void* d_out, int out_size, void* d_ws, size_t ws_size,
                              hipStream_t stream) {
    (void)in_sizes; (void)n_in; (void)out_size; (void)ws_size;
    const float* xs = (const float*)d_in[0];
    const float* attn_q = (const float*)d_in[1];
    const float* attn_k = (const float*)d_in[2];
    const float* attn_v = (const float*)d_in[3];
    const float* attn_o = (const float*)d_in[4];
    const float* attn_r = (const float*)d_in[5];
    const float* mem_f = (const float*)d_in[6];
    const float* mem_q = (const float*)d_in[7];
    const float* mem_k = (const float*)d_in[8];
    const float* mem_v = (const float*)d_in[9];
    const float* mem_o = (const float*)d_in[10];
    const float* mem_r = (const float*)d_in[11];
    const float* mem_beta = (const float*)d_in[12];
    const float* mem_decay = (const float*)d_in[13];
    const float* mem_l = (const float*)d_in[14];
    const float* ff_in1 = (const float*)d_in[15];
    const float* ff_in2 = (const float*)d_in[16];
    const float* ff_out = (const float*)d_in[17];
    const float* sn1_g = (const float*)d_in[18];
    const float* sn2_g = (const float*)d_in[19];
    float* out = (float*)d_out;

    char* base = (char*)d_ws;
    size_t off = 0;
    auto alloc = [&](size_t bytes) -> void* {
        void* p = base + off;
        off += (bytes + 255) & ~(size_t)255;
        return p;
    };
    auto allocF = [&](size_t n) { return (float*)alloc(n * 4); };
    auto allocB = [&](size_t n) { return (bf16*)alloc(n * 2); };

    // bf16 weights
    bf16* w_aq = allocB((size_t)256 * 2048);
    bf16* w_ak = allocB((size_t)256 * 256);
    bf16* w_av = allocB((size_t)2048 * 2048);
    bf16* w_ao = allocB((size_t)2048 * 2048);
    bf16* w_ar = allocB((size_t)2048 * 2048);
    bf16* w_mf = allocB((size_t)64 * 2048);
    bf16* w_mq = allocB((size_t)256 * 2048);
    bf16* w_mv = allocB((size_t)256 * 2048);
    bf16* w_mo = allocB((size_t)2048 * 256);
    bf16* w_mr = allocB((size_t)2048 * 2048);
    bf16* w_mkT = allocB((size_t)256 * 256);
    bf16* w_f1 = allocB((size_t)FFD * 2048);
    bf16* w_f2 = allocB((size_t)FFD * 2048);
    bf16* w_fo = allocB((size_t)2048 * FFD);
    // activations
    bf16* xn = allocB((size_t)T_LEN * XDIM);
    float* qf = allocF((size_t)T_LEN * 256);
    bf16* qb = allocB((size_t)T_LEN * 256);
    float* kf = allocF((size_t)T_LEN * 256);
    float* vf = allocF((size_t)T_LEN * XDIM);
    float* rA = allocF((size_t)T_LEN * XDIM);
    float* qrot = allocF((size_t)T_LEN * 256);
    float* klo = allocF((size_t)T_LEN * 256);
    float* khi = allocF((size_t)T_LEN * 256);
    float* Ab = allocF((size_t)32 * 64 * 128);
    bf16* yb = allocB((size_t)T_LEN * XDIM);
    float* x1 = allocF((size_t)T_LEN * XDIM);
    float* vm = allocF((size_t)T_LEN * 256);
    float* fb = allocF((size_t)T_LEN * 64);
    bf16* memsb = allocB((size_t)T_LEN * 64 * 256);
    float* qmf = allocF((size_t)T_LEN * 256);
    bf16* qmb = allocB((size_t)T_LEN * 256);
    float* qkf = allocF((size_t)T_LEN * 256);
    float* rM = allocF((size_t)T_LEN * XDIM);
    bf16* ymb = allocB((size_t)T_LEN * 256);
    float* x2 = allocF((size_t)T_LEN * XDIM);
    bf16* x2n = allocB((size_t)T_LEN * XDIM);
    float* h1 = allocF((size_t)T_LEN * FFD);
    bf16* hb = allocB((size_t)T_LEN * FFD);

    auto cvt = [&](const float* src, bf16* dst, size_t n) {
        cvt_kernel<<<dim3((unsigned)((n + 255) / 256)), 256, 0, stream>>>(src, dst, n);
    };
    auto gemm = [&](const bf16* A, const bf16* W, float* oF, bf16* oB,
                    const float* gate, const float* res, int M, int N, int K, int mode) {
        if ((N % 128) == 0) {
            dim3 g((unsigned)(N / 128), (unsigned)(M / 128));
            gemm_bf16_kernel<128, 128, 2, 4><<<g, 256, 0, stream>>>(A, W, oF, oB, gate, res, M, N, K, mode);
        } else {
            dim3 g((unsigned)(N / 64), (unsigned)(M / 128));
            gemm_bf16_kernel<128, 64, 4, 2><<<g, 256, 0, stream>>>(A, W, oF, oB, gate, res, M, N, K, mode);
        }
    };

    // weight conversion (fits in L2; done per call, deterministic)
    cvt(attn_q, w_aq, (size_t)256 * 2048);
    cvt(attn_k, w_ak, (size_t)256 * 256);
    cvt(attn_v, w_av, (size_t)2048 * 2048);
    cvt(attn_o, w_ao, (size_t)2048 * 2048);
    cvt(attn_r, w_ar, (size_t)2048 * 2048);
    cvt(mem_f, w_mf, (size_t)64 * 2048);
    cvt(mem_q, w_mq, (size_t)256 * 2048);
    cvt(mem_v, w_mv, (size_t)256 * 2048);
    cvt(mem_o, w_mo, (size_t)2048 * 256);
    cvt(mem_r, w_mr, (size_t)2048 * 2048);
    cvt(ff_in1, w_f1, (size_t)FFD * 2048);
    cvt(ff_in2, w_f2, (size_t)FFD * 2048);
    cvt(ff_out, w_fo, (size_t)2048 * FFD);
    tconv_kernel<<<(256 * 256 + 255) / 256, 256, 0, stream>>>(mem_k, w_mkT, 256, 256);

    // xn = scale_norm(xs, g1)
    norm_kernel<<<T_LEN, 256, 0, stream>>>(xs, sn1_g, xn, XDIM);

    // ---- attention branch ----
    gemm(xn, w_aq, qf, qb, nullptr, nullptr, T_LEN, 256, 2048, 0);      // q
    gemm(qb, w_ak, kf, nullptr, nullptr, nullptr, T_LEN, 256, 256, 0);  // k = q @ Wk^T
    gemm(xn, w_av, vf, nullptr, nullptr, nullptr, T_LEN, 2048, 2048, 0); // v
    gemm(xn, w_ar, rA, nullptr, nullptr, nullptr, T_LEN, 2048, 2048, 1); // sigmoid gate
    rope_kernel<<<(T_LEN * 128 + 255) / 256, 256, 0, stream>>>(qf, kf, qrot, klo, khi);
    attn_scores_kernel<<<32, 256, 0, stream>>>(qrot, klo, khi, Ab);
    attn_y_kernel<<<dim3(8, 32), 256, 0, stream>>>(Ab, vf, yb);
    gemm(yb, w_ao, x1, nullptr, rA, xs, T_LEN, 2048, 2048, 0);          // x1 = xs + attn

    // ---- memory branch ----
    gemm(xn, w_mv, vm, nullptr, nullptr, nullptr, T_LEN, 256, 2048, 0);
    gemm(xn, w_mf, fb, nullptr, nullptr, nullptr, T_LEN, 64, 2048, 1);  // f = sigmoid
    scan_kernel<<<64, 256, 0, stream>>>(fb, vm, mem_beta, mem_decay, memsb);
    gemm(xn, w_mq, qmf, qmb, nullptr, nullptr, T_LEN, 256, 2048, 0);
    gemm(qmb, w_mkT, qkf, nullptr, nullptr, nullptr, T_LEN, 256, 256, 0); // qk = q @ Wk
    mem_attend_kernel<<<T_LEN, 256, 0, stream>>>(memsb, qkf, qmf, mem_l, ymb);
    gemm(xn, w_mr, rM, nullptr, nullptr, nullptr, T_LEN, 2048, 2048, 1);  // sigmoid gate
    gemm(ymb, w_mo, x2, nullptr, rM, x1, T_LEN, 2048, 256, 0);            // x2 = x1 + mem

    // ---- FF block ----
    norm_kernel<<<T_LEN, 256, 0, stream>>>(x2, sn2_g, x2n, XDIM);
    gemm(x2n, w_f1, h1, nullptr, nullptr, nullptr, T_LEN, FFD, 2048, 2);  // silu(h1)
    gemm(x2n, w_f2, nullptr, hb, h1, nullptr, T_LEN, FFD, 2048, 0);       // h = silu(h1)*h2
    gemm(hb, w_fo, out, nullptr, nullptr, x2, T_LEN, 2048, FFD, 0);       // out = h@Wo^T + x2
}
